// Qwen3Attention_31129922962153
// MI455X (gfx1250) — matrix operand order repaired and checked by reading, not compiled, not run on hardware
//
#include <hip/hip_runtime.h>
#include <hip/hip_bf16.h>

// ---------------------------------------------------------------------------
// Qwen3 attention (B=1, S=4096, D=2048, H=16, KVH=2, HD=128) for gfx1250.
// bf16 WMMA (v_wmma_f32_16x16x32_bf16) everywhere; f32 accumulate; f32 softmax.
// Attention K/V tiles staged into LDS by the Tensor Data Mover (TDM).
// ---------------------------------------------------------------------------

#define S_LEN 4096
#define D_DIM 2048
#define NHEAD 16
#define KVHEAD 2
#define HDIM 128
#define QK_DIM (NHEAD * HDIM)   // 2048
#define KV_DIM (KVHEAD * HDIM)  // 256

typedef __bf16 bf16_t;
typedef __attribute__((ext_vector_type(16))) __bf16 v16bf;
typedef __attribute__((ext_vector_type(8)))  __bf16 v8bf;
typedef __attribute__((ext_vector_type(8)))  float  v8f;
typedef __attribute__((ext_vector_type(4)))  unsigned int u32x4;
typedef __attribute__((ext_vector_type(8)))  int i32x8;
typedef __attribute__((ext_vector_type(4)))  int i32x4;

union V16U { v16bf v; v8bf h[2]; };

// A fragment (16x32 bf16, row-major source, row stride ld elements):
// lane<16 : row=lane,    K = {0..7, 16..23}
// lane>=16: row=lane-16, K = {8..15, 24..31}
__device__ __forceinline__ v16bf load_frag_a(const bf16_t* p, int ld, int lane) {
  int half = lane >> 4;
  int row  = lane & 15;
  const bf16_t* r = p + (size_t)row * ld + half * 8;
  V16U f;
  f.h[0] = *(const v8bf*)(r);
  f.h[1] = *(const v8bf*)(r + 16);
  return f.v;
}

// B fragment (32x16 bf16) sourced from a [N][K] (transposed) buffer:
// lane<16 : col=lane,    K = {0..7, 16..23}
// lane>=16: col=lane-16, K = {8..15, 24..31}
__device__ __forceinline__ v16bf load_frag_b(const bf16_t* p, int ld, int lane) {
  int half = lane >> 4;
  int col  = lane & 15;
  const bf16_t* r = p + (size_t)col * ld + half * 8;
  V16U f;
  f.h[0] = *(const v8bf*)(r);
  f.h[1] = *(const v8bf*)(r + 16);
  return f.v;
}

__device__ __forceinline__ v8f wmma_bf16(v16bf a, v16bf b, v8f c) {
  return __builtin_amdgcn_wmma_f32_16x16x32_bf16(false, a, false, b, (short)0, c,
                                                 false, false);
}

// ---------------------------------------------------------------------------
// TDM: issue a 2D tensor tile load Global -> LDS.
// Descriptor bitfields per CDNA5 ISA ch.10.8 (group0 128b, group1 256b).
// data_size = 2 bytes (bf16). tile_d0 = contiguous elements per row,
// tile_d1 = rows, stride_elems = row stride of the global tensor.
// This toolchain exposes the 6-arg builtin:
//   (u32x4 g0, i32x8 g1, i32x4 g2, i32x4 g3, i32x8 extra, i32 cpol)
// ---------------------------------------------------------------------------
__device__ __forceinline__ void tdm_load_2d(unsigned int lds_off, const void* gptr,
                                            unsigned int tile_d0, unsigned int tile_d1,
                                            unsigned long long stride_elems) {
  unsigned long long ga = (unsigned long long)(uintptr_t)gptr;
  u32x4 g0;
  g0.x = 1u;                                   // count=1, user D#, no gather
  g0.y = lds_off;                              // lds_addr [63:32]
  g0.z = (unsigned int)ga;                     // global_addr [95:64]
  g0.w = (unsigned int)(ga >> 32) | (2u << 30);// global_addr hi + type=2
  i32x8 g1;
  g1[0] = (int)(1u << 16);                     // workgroup_mask=0, data_size=1 (2B)
  g1[1] = (int)((tile_d0 & 0xFFFFu) << 16);    // tensor_dim0 = tile_d0 (no OOB)
  g1[2] = (int)((tile_d0 >> 16) | ((tile_d1 & 0xFFFFu) << 16)); // tensor_dim1
  g1[3] = (int)((tile_d1 >> 16) | (tile_d0 << 16));             // tile_dim0
  g1[4] = (int)(tile_d1 & 0xFFFFu);            // tile_dim1 ; tile_dim2=0
  g1[5] = (int)(unsigned int)stride_elems;     // tensor_dim0_stride[31:0]
  g1[6] = (int)(unsigned int)(stride_elems >> 32); // stride[47:32]; dim1_stride=0
  g1[7] = 0;
  i32x4 z4 = {0, 0, 0, 0};
  i32x8 z8 = {0, 0, 0, 0, 0, 0, 0, 0};
  __builtin_amdgcn_tensor_load_to_lds(g0, g1, z4, z4, z8, 0);
}

// ---------------------------------------------------------------------------
// fp32 -> bf16 elementwise convert
// ---------------------------------------------------------------------------
__global__ __launch_bounds__(256) void convert_kernel(const float* __restrict__ src,
                                                      bf16_t* __restrict__ dst,
                                                      int n) {
  int i = blockIdx.x * blockDim.x + threadIdx.x;
  if (i < n) dst[i] = (bf16_t)src[i];
}

// ---------------------------------------------------------------------------
// Transpose-convert: W [K][N] fp32 -> WT [N][K] bf16 (32x32 LDS tiles)
// ---------------------------------------------------------------------------
__global__ __launch_bounds__(256) void transconv_kernel(const float* __restrict__ W,
                                                        bf16_t* __restrict__ WT,
                                                        int K, int N) {
  __shared__ float tile[32][33];
  int kb = blockIdx.y * 32, nb = blockIdx.x * 32;
  int tx = threadIdx.x & 31, ty = threadIdx.x >> 5;  // 32 x 8
#pragma unroll
  for (int i = 0; i < 32; i += 8)
    tile[ty + i][tx] = W[(size_t)(kb + ty + i) * N + nb + tx];
  __syncthreads();
#pragma unroll
  for (int i = 0; i < 32; i += 8)
    WT[(size_t)(nb + ty + i) * K + kb + tx] = (bf16_t)tile[tx][ty + i];
}

// ---------------------------------------------------------------------------
// GEMM: C[M][N] = A[M][K] * BT[N][K]^T (+bias). 8 waves: 2 (M) x 4 (N),
// each wave computes a 16x64 tile via 4 accumulators; K loop in chunks of 32.
// Operands are L2-resident (192MB L2), streamed as coalesced 32B fragments.
// ---------------------------------------------------------------------------
template <bool OUT_F32, bool HAS_BIAS>
__global__ __launch_bounds__(256) void gemm_bf16_kernel(
    const bf16_t* __restrict__ A, const bf16_t* __restrict__ BT,
    const float* __restrict__ bias, void* __restrict__ Cout,
    int M, int N, int K, int ldc) {
  int lane = threadIdx.x & 31;
  int wave = threadIdx.x >> 5;
  int wm = wave & 1;
  int wn = wave >> 1;
  int m0 = blockIdx.y * 32 + wm * 16;
  int n0 = blockIdx.x * 256 + wn * 64;
  if (m0 >= M || n0 >= N) return;  // wave-uniform

  v8f acc[4] = {};
  for (int k0 = 0; k0 < K; k0 += 32) {
    v16bf a = load_frag_a(A + (size_t)m0 * K + k0, K, lane);
#pragma unroll
    for (int t = 0; t < 4; ++t) {
      v16bf b = load_frag_b(BT + (size_t)(n0 + t * 16) * K + k0, K, lane);
      acc[t] = wmma_bf16(a, b, acc[t]);
    }
  }

  int col = lane & 15, half = lane >> 4;
#pragma unroll
  for (int t = 0; t < 4; ++t) {
    float bv = 0.0f;
    if (HAS_BIAS) bv = bias[n0 + t * 16 + col];
#pragma unroll
    for (int r = 0; r < 8; ++r) {
      size_t row = (size_t)(m0 + half * 8 + r);
      int cc = n0 + t * 16 + col;
      float v = acc[t][r] + bv;
      if (OUT_F32)
        ((float*)Cout)[row * ldc + cc] = v;
      else
        ((bf16_t*)Cout)[row * ldc + cc] = (bf16_t)v;
    }
  }
}

// ---------------------------------------------------------------------------
// RoPE (in-place on bf16 Q and K). One thread per (s, head, d<64) pair.
// ---------------------------------------------------------------------------
__global__ __launch_bounds__(256) void rope_kernel(bf16_t* __restrict__ Q,
                                                   bf16_t* __restrict__ Kb,
                                                   const int* __restrict__ pos_ids) {
  int idx = blockIdx.x * blockDim.x + threadIdx.x;
  int d = idx & 63;
  int t = idx >> 6;
  int head = t % (NHEAD + KVHEAD);
  int s = t / (NHEAD + KVHEAD);
  if (s >= S_LEN) return;
  float pos = (float)pos_ids[s];
  // inv_freq = theta^(-d/64), theta = 1e6 ; log(1e6) = 13.815511
  float freq = __expf(-(float)d * (1.0f / 64.0f) * 13.815510557964274f);
  float ang = pos * freq;
  float c = __cosf(ang), sn = __sinf(ang);
  bf16_t* buf;
  size_t off;
  if (head < NHEAD) {
    buf = Q;
    off = (size_t)s * QK_DIM + head * HDIM + d;
  } else {
    buf = Kb;
    off = (size_t)s * KV_DIM + (head - NHEAD) * HDIM + d;
  }
  float x1 = (float)buf[off];
  float x2 = (float)buf[off + 64];
  buf[off]      = (bf16_t)(x1 * c - x2 * sn);
  buf[off + 64] = (bf16_t)(x2 * c + x1 * sn);
}

// ---------------------------------------------------------------------------
// V transpose: V [S][KV_DIM] -> Vt [KV_DIM][S]  (bf16)
// ---------------------------------------------------------------------------
__global__ __launch_bounds__(256) void vtrans_kernel(const bf16_t* __restrict__ V,
                                                     bf16_t* __restrict__ Vt) {
  int idx = blockIdx.x * blockDim.x + threadIdx.x;
  int c = idx & (KV_DIM - 1);
  int s = idx >> 8;
  if (s < S_LEN) Vt[(size_t)c * S_LEN + s] = V[(size_t)s * KV_DIM + c];
}

// ---------------------------------------------------------------------------
// Flash attention. Block = (head, 128-query chunk): 8 waves x 16 queries.
// Per 32-key block: wave 0 issues two TDM tensor loads (K tile 32x128 and
// Vt tile 128x32 -> LDS), waits TENSORcnt, block barriers; all waves then
// read WMMA fragments from LDS. Online softmax in f32; P is reshaped from
// C layout to an A fragment through a per-wave LDS scratch.
// Dynamic LDS: [0,8K) K tile, [8K,16K) V tile, [16K,24K) P scratch (1K/wave).
// ---------------------------------------------------------------------------
__global__ __launch_bounds__(256) void attn_kernel(const bf16_t* __restrict__ Q,
                                                   const bf16_t* __restrict__ Kb,
                                                   const bf16_t* __restrict__ Vt,
                                                   bf16_t* __restrict__ O) {
  extern __shared__ char smem[];
  bf16_t* Kt = (bf16_t*)smem;              // [32][128]
  bf16_t* Vl = (bf16_t*)(smem + 8192);     // [128][32]
  int lane = threadIdx.x & 31;
  int wave = threadIdx.x >> 5;
  bf16_t* pw = (bf16_t*)(smem + 16384) + wave * 512;  // [16][32] per wave

  int head = blockIdx.x >> 5;              // 32 query-chunks per head
  int chunk = blockIdx.x & 31;
  int q0 = chunk * 128 + wave * 16;
  int kvh = head >> 3;                     // GQA group of 8
  const float scale = 0.08838834764831845f;  // 1/sqrt(128)

  // Q fragments for the 4 k=32 chunks of HD=128 (reused for all key blocks)
  v16bf qf[4];
#pragma unroll
  for (int c = 0; c < 4; ++c)
    qf[c] = load_frag_a(Q + (size_t)q0 * QK_DIM + head * HDIM + c * 32, QK_DIM, lane);

  v8f o[8] = {};
  float m[8], l[8];
#pragma unroll
  for (int r = 0; r < 8; ++r) { m[r] = -1.0e30f; l[r] = 0.0f; }

  int half = lane >> 4, col = lane & 15;

  for (int j = 0; j < S_LEN; j += 32) {
    __syncthreads();  // previous iteration's LDS consumers done
    if (wave == 0) {  // wave-uniform; TDM ignores EXEC
      // K tile: 32 key rows x 128 dims from Kb [S][KV_DIM]
      tdm_load_2d(0u, Kb + (size_t)j * KV_DIM + kvh * HDIM, 128u, 32u,
                  (unsigned long long)KV_DIM);
      // V tile: 128 dim rows x 32 keys from Vt [KV_DIM][S]
      tdm_load_2d(8192u, Vt + (size_t)(kvh * HDIM) * S_LEN + j, 32u, 128u,
                  (unsigned long long)S_LEN);
      __builtin_amdgcn_s_wait_tensorcnt(0);
    }
    __syncthreads();  // tiles visible to all waves

    // ---- scores: 16 queries x 32 keys, inner dim HD=128 -> 8 WMMA ----
    v8f s0 = {}, s1 = {};
#pragma unroll
    for (int c = 0; c < 4; ++c) {
      v16bf b0 = load_frag_b(Kt + 0 * 128 + c * 32, 128, lane);
      v16bf b1 = load_frag_b(Kt + 16 * 128 + c * 32, 128, lane);
      s0 = wmma_bf16(qf[c], b0, s0);
      s1 = wmma_bf16(qf[c], b1, s1);
    }

    // ---- online softmax (rows split: VGPR r holds row r + 8*half) ----
    float corr[8];
#pragma unroll
    for (int r = 0; r < 8; ++r) {
      float a0 = s0[r] * scale;
      float a1 = s1[r] * scale;
      float mx = fmaxf(a0, a1);
#pragma unroll
      for (int off = 1; off < 16; off <<= 1)
        mx = fmaxf(mx, __shfl_xor(mx, off, 32));
      float mn = fmaxf(m[r], mx);
      corr[r] = __expf(m[r] - mn);
      m[r] = mn;
      float p0 = __expf(a0 - mn);
      float p1 = __expf(a1 - mn);
      float rs = p0 + p1;
#pragma unroll
      for (int off = 1; off < 16; off <<= 1)
        rs += __shfl_xor(rs, off, 32);
      l[r] = l[r] * corr[r] + rs;
      // stash P (bf16) in per-wave LDS scratch at [row][key-in-block]
      pw[(half * 8 + r) * 32 + col]      = (bf16_t)p0;
      pw[(half * 8 + r) * 32 + col + 16] = (bf16_t)p1;
    }

    // rescale running output accumulators
#pragma unroll
    for (int n = 0; n < 8; ++n)
#pragma unroll
      for (int r = 0; r < 8; ++r) o[n][r] *= corr[r];

    // intra-wave LDS RAW: wait for stores, then load P as an A fragment
    asm volatile("s_wait_dscnt 0" ::: "memory");
    v16bf pf = load_frag_a(pw, 32, lane);

    // ---- O += P (16x32) * V (32x128): 8 WMMA over HD tiles ----
#pragma unroll
    for (int n = 0; n < 8; ++n) {
      v16bf bv = load_frag_b(Vl + (size_t)(n * 16) * 32, 32, lane);
      o[n] = wmma_bf16(pf, bv, o[n]);
    }
  }

  // normalize and write out (bf16, [S][H*HD])
#pragma unroll
  for (int r = 0; r < 8; ++r) l[r] = 1.0f / l[r];
#pragma unroll
  for (int n = 0; n < 8; ++n)
#pragma unroll
    for (int r = 0; r < 8; ++r) {
      size_t row = (size_t)(q0 + half * 8 + r);
      O[row * QK_DIM + head * HDIM + n * 16 + col] = (bf16_t)(o[n][r] * l[r]);
    }
}

// ---------------------------------------------------------------------------
// Host launcher
// ---------------------------------------------------------------------------
extern "C" void kernel_launch(void* const* d_in, const int* in_sizes, int n_in,
                              void* d_out, int out_size, void* d_ws, size_t ws_size,
                              hipStream_t stream) {
  (void)in_sizes; (void)n_in; (void)out_size; (void)ws_size;
  const float* hidden = (const float*)d_in[0];
  const int*   pos    = (const int*)d_in[1];
  const float* Wq     = (const float*)d_in[2];
  const float* bq     = (const float*)d_in[3];
  const float* Wk     = (const float*)d_in[4];
  const float* bk     = (const float*)d_in[5];
  const float* Wv     = (const float*)d_in[6];
  const float* bv     = (const float*)d_in[7];
  const float* Wo     = (const float*)d_in[8];
  float* out = (float*)d_out;

  // workspace carve-up (bf16 buffers)
  char* ws = (char*)d_ws;
  bf16_t* hB  = (bf16_t*)ws; ws += (size_t)S_LEN * D_DIM * 2;     // 16 MB
  bf16_t* WqT = (bf16_t*)ws; ws += (size_t)QK_DIM * D_DIM * 2;    //  8 MB
  bf16_t* WkT = (bf16_t*)ws; ws += (size_t)KV_DIM * D_DIM * 2;    //  1 MB
  bf16_t* WvT = (bf16_t*)ws; ws += (size_t)KV_DIM * D_DIM * 2;    //  1 MB
  bf16_t* WoT = (bf16_t*)ws; ws += (size_t)D_DIM * QK_DIM * 2;    //  8 MB
  bf16_t* QB  = (bf16_t*)ws; ws += (size_t)S_LEN * QK_DIM * 2;    // 16 MB
  bf16_t* KB  = (bf16_t*)ws; ws += (size_t)S_LEN * KV_DIM * 2;    //  2 MB
  bf16_t* VB  = (bf16_t*)ws; ws += (size_t)S_LEN * KV_DIM * 2;    //  2 MB
  bf16_t* VtB = (bf16_t*)ws; ws += (size_t)KV_DIM * S_LEN * 2;    //  2 MB
  bf16_t* AT  = (bf16_t*)ws; ws += (size_t)S_LEN * QK_DIM * 2;    // 16 MB

  // 1) precision conversion + weight transposes
  {
    int n = S_LEN * D_DIM;
    convert_kernel<<<(n + 255) / 256, 256, 0, stream>>>(hidden, hB, n);
  }
  transconv_kernel<<<dim3(QK_DIM / 32, D_DIM / 32), 256, 0, stream>>>(Wq, WqT, D_DIM, QK_DIM);
  transconv_kernel<<<dim3(KV_DIM / 32, D_DIM / 32), 256, 0, stream>>>(Wk, WkT, D_DIM, KV_DIM);
  transconv_kernel<<<dim3(KV_DIM / 32, D_DIM / 32), 256, 0, stream>>>(Wv, WvT, D_DIM, KV_DIM);
  transconv_kernel<<<dim3(D_DIM / 32, QK_DIM / 32), 256, 0, stream>>>(Wo, WoT, QK_DIM, D_DIM);

  // 2) QKV projections (WMMA GEMMs, bias fused)
  gemm_bf16_kernel<false, true><<<dim3(QK_DIM / 256, S_LEN / 32), 256, 0, stream>>>(
      hB, WqT, bq, QB, S_LEN, QK_DIM, D_DIM, QK_DIM);
  gemm_bf16_kernel<false, true><<<dim3(1, S_LEN / 32), 256, 0, stream>>>(
      hB, WkT, bk, KB, S_LEN, KV_DIM, D_DIM, KV_DIM);
  gemm_bf16_kernel<false, true><<<dim3(1, S_LEN / 32), 256, 0, stream>>>(
      hB, WvT, bv, VB, S_LEN, KV_DIM, D_DIM, KV_DIM);

  // 3) RoPE on Q and K; transpose V for the P*V WMMA layout
  {
    int n = S_LEN * (NHEAD + KVHEAD) * 64;
    rope_kernel<<<(n + 255) / 256, 256, 0, stream>>>(QB, KB, pos);
    int m = S_LEN * KV_DIM;
    vtrans_kernel<<<(m + 255) / 256, 256, 0, stream>>>(VB, VtB);
  }

  // 4) flash attention: 512 blocks (16 heads x 32 query chunks), 24KB LDS
  attn_kernel<<<dim3(NHEAD * 32), 256, 24576, stream>>>(QB, KB, VtB, AT);

  // 5) output projection -> fp32 d_out
  gemm_bf16_kernel<true, false><<<dim3(D_DIM / 256, S_LEN / 32), 256, 0, stream>>>(
      AT, WoT, nullptr, out, S_LEN, D_DIM, QK_DIM, D_DIM);
}
